// DSARF_5875515261033
// MI455X (gfx1250) — matrix operand order repaired and checked by reading, not compiled, not run on hardware
//
#include <hip/hip_runtime.h>

#define T_    500
#define B_    64
#define D_    256
#define Z_    64
#define H_    128
#define S_    8
#define LN_   3
#define TD_   64
#define MAXL_ 5
#define ALPHA_ 0.25f

typedef __attribute__((ext_vector_type(16))) _Float16 v16h;
typedef __attribute__((ext_vector_type(8)))  _Float16 v8h;
typedef __attribute__((ext_vector_type(8)))  float    v8f;

__device__ __forceinline__ float preluf_(float x) { return x >= 0.f ? x : ALPHA_ * x; }
__device__ __forceinline__ float sigmf_(float x)  { return 1.f / (1.f + __expf(-x)); }

// ---------------- CDNA5 wave32 WMMA fragments, fed straight from memory ----------------
// A (16x32 MxK) f16, row-major with stride lda: lane's data = two contiguous 8-half runs.
__device__ __forceinline__ v16h frag_a_g(const _Float16* __restrict__ A, int lda,
                                         int m0, int lane, int kc) {
  const _Float16* row = A + (size_t)(m0 + (lane & 15)) * lda + kc + ((lane >= 16) ? 8 : 0);
  v8h lo = *(const v8h*)(row);
  v8h hi = *(const v8h*)(row + 16);
  return __builtin_shufflevector(lo, hi, 0,1,2,3,4,5,6,7,8,9,10,11,12,13,14,15);
}
// B (32x16 KxN) from an [N,K] row-major f16 operand: lane's data = two contiguous 8-half runs.
__device__ __forceinline__ v16h frag_b_g(const _Float16* __restrict__ Bt, int ldb,
                                         int n0, int lane, int kc) {
  const _Float16* row = Bt + (size_t)(n0 + (lane & 15)) * ldb + kc + ((lane >= 16) ? 8 : 0);
  v8h lo = *(const v8h*)(row);
  v8h hi = *(const v8h*)(row + 16);
  return __builtin_shufflevector(lo, hi, 0,1,2,3,4,5,6,7,8,9,10,11,12,13,14,15);
}
// A fragment from an LDS-resident f32 matrix, with optional scale / PReLU fused in.
__device__ __forceinline__ v16h frag_a_l(const float* __restrict__ A, int lda,
                                         int m0, int lane, int kc, float scale, int act) {
  const float* row = A + (m0 + (lane & 15)) * lda + kc + ((lane >= 16) ? 8 : 0);
  v16h a;
#pragma unroll
  for (int i = 0; i < 8; ++i) {
    float v = scale * row[i]; if (act) v = preluf_(v);
    a[i] = (_Float16)v;
  }
#pragma unroll
  for (int i = 0; i < 8; ++i) {
    float v = scale * row[16 + i]; if (act) v = preluf_(v);
    a[8 + i] = (_Float16)v;
  }
  return a;
}

// 256 threads = 8 waves; wave w owns 16x16 tiles ti = w and w+8 of a 64x64 output
// window [0..64) x [bn0..bn0+64); accumulates f32 into acc[0..1].
__device__ __forceinline__ void gemm_gg(const _Float16* __restrict__ A, int lda,
                                        const _Float16* __restrict__ Bt, int ldb, int bn0,
                                        int K, v8f* acc, int wave, int lane) {
  for (int kc = 0; kc < K; kc += 32) {
#pragma unroll
    for (int e = 0; e < 2; ++e) {
      int ti = wave + e * 8;
      v16h af = frag_a_g(A, lda, (ti >> 2) * 16, lane, kc);
      v16h bf = frag_b_g(Bt, ldb, bn0 + (ti & 3) * 16, lane, kc);
      acc[e] = __builtin_amdgcn_wmma_f32_16x16x32_f16(false, af, false, bf,
                                                      (short)0, acc[e], false, false);
    }
  }
}
__device__ __forceinline__ void gemm_lg(const float* __restrict__ A, int lda, float scale, int act,
                                        const _Float16* __restrict__ Bt, int ldb, int bn0,
                                        int K, v8f* acc, int wave, int lane) {
  for (int kc = 0; kc < K; kc += 32) {
#pragma unroll
    for (int e = 0; e < 2; ++e) {
      int ti = wave + e * 8;
      v16h af = frag_a_l(A, lda, (ti >> 2) * 16, lane, kc, scale, act);
      v16h bf = frag_b_g(Bt, ldb, bn0 + (ti & 3) * 16, lane, kc);
      acc[e] = __builtin_amdgcn_wmma_f32_16x16x32_f16(false, af, false, bf,
                                                      (short)0, acc[e], false, false);
    }
  }
}

// C/D layout: VGPR r -> M = 16*mt + r + (lane>=16?8:0), N = 16*nt + (lane&15).
#define TILE_FOREACH(ACC, ...)                                   \
  for (int e_ = 0; e_ < 2; ++e_) {                               \
    int ti_ = wave + e_ * 8;                                     \
    int mb_ = ((ti_ >> 2) << 4) + ((lane >= 16) ? 8 : 0);        \
    int n   = ((ti_ & 3) << 4) + (lane & 15);                    \
    for (int r_ = 0; r_ < 8; ++r_) {                             \
      int m = mb_ + r_;                                          \
      float vacc = ACC[e_][r_];                                  \
      __VA_ARGS__                                                \
    }                                                            \
  }

// ---------------- one-time operand prep: f32 -> f16, optional transpose + row padding ----
__global__ void cvt_f16(const float* __restrict__ src, _Float16* __restrict__ dst, int n) {
  for (int i = (int)blockIdx.x * (int)blockDim.x + (int)threadIdx.x; i < n;
       i += (int)gridDim.x * (int)blockDim.x)
    dst[i] = (_Float16)src[i];
}
// dst[b][n][k] (n < padN): n<N ? (trans ? src[b][k][n] : src[b][n][k]) : 0
__global__ void prep_w(const float* __restrict__ src, _Float16* __restrict__ dst,
                       int batch, int N, int K, int padN, int trans) {
  int total = batch * padN * K;
  for (int i = (int)blockIdx.x * (int)blockDim.x + (int)threadIdx.x; i < total;
       i += (int)gridDim.x * (int)blockDim.x) {
    int b = i / (padN * K);
    int r = i - b * padN * K;
    int n = r / K, k = r - n * K;
    float v = 0.f;
    if (n < N) v = trans ? src[(size_t)b * K * N + (size_t)k * N + n]
                         : src[(size_t)b * N * K + (size_t)n * K + k];
    dst[i] = (_Float16)v;
  }
}

// ---------------- big parallel GEMM: C = act(A@Bt^T(+bias1+bias2)), f32 and/or f16 out ----
__global__ __launch_bounds__(256) void gemm_big(
    const _Float16* __restrict__ A, int lda,
    const _Float16* __restrict__ Bt, int ldb, int Nd,
    const float* __restrict__ bias1, const float* __restrict__ bias2,
    float* __restrict__ Cf, _Float16* __restrict__ Ch, int ldc, int K, int act)
{
  const int wave = (int)threadIdx.x >> 5, lane = (int)threadIdx.x & 31;
  const _Float16* Ab = A + (size_t)blockIdx.x * 64 * lda;
  const int bn0 = (int)blockIdx.y * 64;
  v8f acc[2] = {};
  gemm_gg(Ab, lda, Bt, ldb, bn0, K, acc, wave, lane);
  TILE_FOREACH(acc,
    int gn = bn0 + n;
    if (gn < Nd) {
      float v = vacc;
      if (bias1) v += bias1[gn];
      if (bias2) v += bias2[gn];
      if (act) v = preluf_(v);
      size_t off = (size_t)((size_t)blockIdx.x * 64 + m) * ldc + gn;
      if (Cf) Cf[off] = v;
      if (Ch) Ch[off] = (_Float16)v;
    }
  )
}

// ---------------- persistent single-WGP LSTM recurrence (gates i,f,g,o) ----------------
__global__ __launch_bounds__(256) void lstm_seq(
    const float* __restrict__ Xp,        // [T,B,4H], already includes bih+bhh
    const _Float16* __restrict__ Whh16,  // [4H,H] f16
    _Float16* __restrict__ hs16,         // [T,B,H] f16 out
    _Float16* __restrict__ hSt16, float* __restrict__ cSt, float* __restrict__ gBuf)
{
  const int tid = (int)threadIdx.x;
  const int wave = tid >> 5, lane = tid & 31;
  for (int i = tid; i < B_ * H_; i += 256) { hSt16[i] = (_Float16)0.f; cSt[i] = 0.f; }
  __syncthreads();
  for (int t = 0; t < T_; ++t) {
    const float* xp = Xp + (size_t)t * B_ * 4 * H_;
    for (int nc = 0; nc < 8; ++nc) {
      v8f acc[2] = {};
      gemm_gg(hSt16, H_, Whh16, H_, nc * 64, H_, acc, wave, lane);
      TILE_FOREACH(acc,
        int gn = nc * 64 + n;
        gBuf[(size_t)m * 512 + gn] = vacc + xp[(size_t)m * 512 + gn];
      )
    }
    __syncthreads();
    for (int i = tid; i < B_ * H_; i += 256) {
      int b = i >> 7, j = i & 127;
      const float* g = gBuf + (size_t)b * 512;
      float c = sigmf_(g[128 + j]) * cSt[i] + sigmf_(g[j]) * tanhf(g[256 + j]);
      float h = sigmf_(g[384 + j]) * tanhf(c);
      cSt[i] = c;
      hSt16[i] = (_Float16)h;
      hs16[(size_t)t * B_ * H_ + i] = (_Float16)h;
    }
    __syncthreads();
  }
}

// ---------------- lag-ring init (f16): slot j = z0_mu[j] broadcast over batch ----------
__global__ void init_buf(const float* __restrict__ z0, _Float16* __restrict__ buf) {
  int i = (int)blockIdx.x * (int)blockDim.x + (int)threadIdx.x;
  if (i < MAXL_ * B_ * Z_) {
    int j = i / (B_ * Z_);
    int z = i & (Z_ - 1);
    buf[i] = (_Float16)z0[j * Z_ + z];
  }
}

// ---------------- per-step parallel part: gated transition (blocks 0..7) + state (8) ----
__global__ __launch_bounds__(256) void step_par(int t,
    const _Float16* __restrict__ ring, const float* __restrict__ sq,
    const _Float16* __restrict__ g1gT, const float* __restrict__ g1gb,
    const _Float16* __restrict__ g2gT, const float* __restrict__ g2gb,
    const _Float16* __restrict__ g1zT, const float* __restrict__ g1zb,
    const _Float16* __restrict__ g2zT, const float* __restrict__ g2zb,
    const _Float16* __restrict__ g3zT, const float* __restrict__ g3zb,
    const _Float16* __restrict__ gzT,  const float* __restrict__ gzb,
    const _Float16* __restrict__ st1_16, const float* __restrict__ st1_b,
    const _Float16* __restrict__ st2_16, const float* __restrict__ st2_b,
    const float* __restrict__ st3_w, const float* __restrict__ st3_b,
    float* __restrict__ pmu, float* __restrict__ psig,
    float* __restrict__ qs, float* __restrict__ ps)
{
  __shared__ float accA[64 * 64];
  __shared__ float accB[64 * 64];
  __shared__ float gateB[64 * 64];
  const int tid = (int)threadIdx.x;
  const int wave = tid >> 5, lane = tid & 31;
  const int s = (int)blockIdx.x;
  const int lags[3] = {1, 2, 5};
  const _Float16* zl[3];
  for (int i = 0; i < 3; ++i) {
    int ph = ((t - lags[i]) % 5 + 5) % 5;
    zl[i] = ring + (size_t)ph * B_ * Z_;
  }

  if (s < S_) {
    __builtin_prefetch(g1gT + (size_t)s * LN_ * Z_ * TD_, 0, 0);
    __builtin_prefetch(g1zT + (size_t)s * LN_ * Z_ * TD_, 0, 0);
    __builtin_prefetch(gzT + (size_t)s * LN_ * Z_ * Z_, 0, 0);
    for (int i = tid; i < 64 * 64; i += 256) { accA[i] = 0.f; accB[i] = 0.f; }
    __syncthreads();
    // Ag = sum_l prelu(z_lag[l]@g1g[s,l] + g1gb[s,l]); Am likewise with g1z
    for (int l = 0; l < LN_; ++l) {
      {
        v8f acc[2] = {};
        gemm_gg(zl[l], Z_, g1gT + (size_t)(s * LN_ + l) * TD_ * Z_, Z_, 0, Z_, acc, wave, lane);
        const float* bb = g1gb + (size_t)(s * LN_ + l) * TD_;
        TILE_FOREACH(acc, accA[m * 64 + n] += preluf_(vacc + bb[n]);)
      }
      {
        v8f acc[2] = {};
        gemm_gg(zl[l], Z_, g1zT + (size_t)(s * LN_ + l) * TD_ * Z_, Z_, 0, Z_, acc, wave, lane);
        const float* bb = g1zb + (size_t)(s * LN_ + l) * TD_;
        TILE_FOREACH(acc, accB[m * 64 + n] += preluf_(vacc + bb[n]);)
      }
    }
    __syncthreads();
    // gate = sigmoid(mean(Ag)@g2g[s] + g2gb[s])
    {
      v8f acc[2] = {};
      gemm_lg(accA, 64, 1.f / 3.f, 0, g2gT + (size_t)s * Z_ * TD_, TD_, 0, TD_, acc, wave, lane);
      const float* bb = g2gb + (size_t)s * Z_;
      TILE_FOREACH(acc, gateB[m * 64 + n] = sigmf_(vacc + bb[n]);)
    }
    __syncthreads();
    // z_mean = mean(Am)@g2z[s] + g2zb[s]  (into accA; Ag no longer needed)
    {
      v8f acc[2] = {};
      gemm_lg(accB, 64, 1.f / 3.f, 0, g2zT + (size_t)s * Z_ * TD_, TD_, 0, TD_, acc, wave, lane);
      const float* bb = g2zb + (size_t)s * Z_;
      TILE_FOREACH(acc, accA[m * 64 + n] = vacc + bb[n];)
    }
    __syncthreads();
    // z_lin = mean_l(z_lag[l]@gz[s,l] + gzb[s,l]);  p_mu = (1-gate)*z_lin + gate*z_mean
    {
      v8f acc[2] = {};
      for (int l = 0; l < LN_; ++l)
        gemm_gg(zl[l], Z_, gzT + (size_t)(s * LN_ + l) * Z_ * Z_, Z_, 0, Z_, acc, wave, lane);
      TILE_FOREACH(acc,
        float bz = gzb[(size_t)(s * LN_ + 0) * Z_ + n]
                 + gzb[(size_t)(s * LN_ + 1) * Z_ + n]
                 + gzb[(size_t)(s * LN_ + 2) * Z_ + n];
        float zlin = (vacc + bz) * (1.f / 3.f);
        float g = gateB[m * 64 + n];
        float zm = accA[m * 64 + n];
        pmu[(((size_t)t * S_ + s) * B_ + m) * Z_ + n] = (1.f - g) * zlin + g * zm;
      )
    }
    __syncthreads();
    // p_sig = prelu(z_mean)@g3z[s] + g3zb[s]  (PReLU fused into A fragment load)
    {
      v8f acc[2] = {};
      gemm_lg(accA, 64, 1.f, 1, g3zT + (size_t)s * Z_ * Z_, Z_, 0, Z_, acc, wave, lane);
      const float* bb = g3zb + (size_t)s * Z_;
      TILE_FOREACH(acc, psig[(((size_t)t * S_ + s) * B_ + m) * Z_ + n] = vacc + bb[n];)
    }
  } else {
    // state transition p(s_t | z_{t-1}) + q_s softmax
    int ph = ((t - 1) % 5 + 5) % 5;
    const _Float16* zprev = ring + (size_t)ph * B_ * Z_;
    {
      v8f acc[2] = {};
      gemm_gg(zprev, Z_, st1_16, Z_, 0, Z_, acc, wave, lane);
      TILE_FOREACH(acc, accA[m * 64 + n] = preluf_(vacc + st1_b[n]);)
    }
    __syncthreads();
    {
      v8f acc[2] = {};
      gemm_lg(accA, 64, 1.f, 0, st2_16, Z_, 0, Z_, acc, wave, lane);
      TILE_FOREACH(acc, accB[m * 64 + n] = preluf_(vacc + st2_b[n]);)
    }
    __syncthreads();
    if (tid < B_) {
      int b = tid;
      float lg[S_];
      float mx = -1e30f;
      for (int n2 = 0; n2 < S_; ++n2) {
        float d = st3_b[n2];
        for (int k = 0; k < Z_; ++k) d += accB[b * 64 + k] * st3_w[n2 * Z_ + k];
        lg[n2] = d; mx = fmaxf(mx, d);
      }
      float sum = 0.f;
      for (int n2 = 0; n2 < S_; ++n2) { lg[n2] = __expf(lg[n2] - mx); sum += lg[n2]; }
      for (int n2 = 0; n2 < S_; ++n2)
        ps[((size_t)t * B_ + b) * S_ + n2] = lg[n2] / sum;
      const float* sl = sq + ((size_t)t * B_ + b) * S_;
      mx = -1e30f;
      for (int n2 = 0; n2 < S_; ++n2) mx = fmaxf(mx, sl[n2]);
      sum = 0.f;
      for (int n2 = 0; n2 < S_; ++n2) { lg[n2] = __expf(sl[n2] - mx); sum += lg[n2]; }
      for (int n2 = 0; n2 < S_; ++n2)
        qs[((size_t)t * B_ + b) * S_ + n2] = lg[n2] / sum;
    }
  }
}

// ---------------- per-step combiner + emission + lag-ring update (1 block) ----------------
__global__ __launch_bounds__(256) void step_comb(int t,
    _Float16* __restrict__ ring, const _Float16* __restrict__ zq16,
    const _Float16* __restrict__ c1T, const float* __restrict__ c1b,
    const _Float16* __restrict__ c2_16, const float* __restrict__ c2_b,
    const _Float16* __restrict__ c21_16, const float* __restrict__ c21_b,
    const _Float16* __restrict__ c3_16, const float* __restrict__ c3_b,
    const _Float16* __restrict__ c31_16, const float* __restrict__ c31_b,
    const _Float16* __restrict__ ew_16, const float* __restrict__ e_b,
    float* __restrict__ yout, float* __restrict__ qzmu, float* __restrict__ qzsig)
{
  __shared__ float hcA[64 * 128];  // prelu(h_rnn) + mean_l prelu(hh_l); hc = 0.5*hcA
  __shared__ float uB[64 * 64];
  const int tid = (int)threadIdx.x;
  const int wave = tid >> 5, lane = tid & 31;
  const int lags[3] = {1, 2, 5};
  const _Float16* zl[3];
  for (int i = 0; i < 3; ++i) {
    int ph = ((t - lags[i]) % 5 + 5) % 5;
    zl[i] = ring + (size_t)ph * B_ * Z_;
  }
  __builtin_prefetch(c1T, 0, 0);
  __builtin_prefetch(ew_16, 0, 0);
  const _Float16* hr = zq16 + (size_t)t * B_ * 128;
  for (int i = tid; i < 64 * 128; i += 256) hcA[i] = preluf_((float)hr[i]);
  __syncthreads();
  // hh_l = z_lag[l]@c1[l] + c1b[l];  hcA += mean_l prelu(hh_l)   (c1T: [l][128n][64k])
  for (int l = 0; l < LN_; ++l) {
    for (int half = 0; half < 2; ++half) {
      v8f acc[2] = {};
      gemm_gg(zl[l], Z_, c1T + (size_t)l * 128 * Z_, Z_, half * 64, Z_, acc, wave, lane);
      const float* bb = c1b + (size_t)l * 128;
      TILE_FOREACH(acc,
        int gn = half * 64 + n;
        hcA[m * 128 + gn] += (1.f / 3.f) * preluf_(vacc + bb[gn]);
      )
    }
  }
  __syncthreads();
  // u = prelu(hc@c2^T + c2_b)   (hc = 0.5*hcA fused via scale)
  {
    v8f acc[2] = {};
    gemm_lg(hcA, 128, 0.5f, 0, c2_16, 128, 0, 128, acc, wave, lane);
    TILE_FOREACH(acc, uB[m * 64 + n] = preluf_(vacc + c2_b[n]);)
  }
  __syncthreads();
  // loc = u@c21^T + c21_b -> q_z_mu[t] (f32) and lag-ring slot t%5 (f16)
  _Float16* bslot = ring + (size_t)(t % 5) * B_ * Z_;
  {
    v8f acc[2] = {};
    gemm_lg(uB, 64, 1.f, 0, c21_16, Z_, 0, Z_, acc, wave, lane);
    TILE_FOREACH(acc,
      float v = vacc + c21_b[n];
      qzmu[(size_t)t * B_ * Z_ + (size_t)m * Z_ + n] = v;
      bslot[m * Z_ + n] = (_Float16)v;
    )
  }
  __syncthreads();
  // u2 = prelu(hc@c3^T + c3_b)
  {
    v8f acc[2] = {};
    gemm_lg(hcA, 128, 0.5f, 0, c3_16, 128, 0, 128, acc, wave, lane);
    TILE_FOREACH(acc, uB[m * 64 + n] = preluf_(vacc + c3_b[n]);)
  }
  __syncthreads();
  // scale = u2@c31^T + c31_b -> q_z_sig[t]
  {
    v8f acc[2] = {};
    gemm_lg(uB, 64, 1.f, 0, c31_16, Z_, 0, Z_, acc, wave, lane);
    TILE_FOREACH(acc, qzsig[(size_t)t * B_ * Z_ + (size_t)m * Z_ + n] = vacc + c31_b[n];)
  }
  __threadfence();
  __syncthreads();
  // emission y = loc@e_w^T + e_b  (A = freshly written f16 ring slot)
  for (int q = 0; q < 4; ++q) {
    v8f acc[2] = {};
    gemm_gg(bslot, Z_, ew_16, Z_, q * 64, Z_, acc, wave, lane);
    TILE_FOREACH(acc,
      int gn = q * 64 + n;
      yout[(size_t)t * B_ * D_ + (size_t)m * D_ + gn] = vacc + e_b[gn];
    )
  }
}

// ---------------- host driver ----------------
extern "C" void kernel_launch(void* const* d_in, const int* in_sizes, int n_in,
                              void* d_out, int out_size, void* d_ws, size_t ws_size,
                              hipStream_t stream)
{
  (void)in_sizes; (void)n_in; (void)out_size; (void)ws_size;
  const float* x    = (const float*)d_in[0];
  const float* Wih0 = (const float*)d_in[1];
  const float* Whh0 = (const float*)d_in[2];
  const float* bih0 = (const float*)d_in[3];
  const float* bhh0 = (const float*)d_in[4];
  const float* Wih1 = (const float*)d_in[5];
  const float* Whh1 = (const float*)d_in[6];
  const float* bih1 = (const float*)d_in[7];
  const float* bhh1 = (const float*)d_in[8];
  const float* f1_w = (const float*)d_in[9];  const float* f1_b = (const float*)d_in[10];
  const float* f2_w = (const float*)d_in[11]; const float* f2_b = (const float*)d_in[12];
  const float* f3_w = (const float*)d_in[13]; const float* f3_b = (const float*)d_in[14];
  const float* s1_w = (const float*)d_in[15]; const float* s1_b = (const float*)d_in[16];
  const float* s2_w = (const float*)d_in[17]; const float* s2_b = (const float*)d_in[18];
  const float* s3_w = (const float*)d_in[19]; const float* s3_b = (const float*)d_in[20];
  const float* c1   = (const float*)d_in[21]; const float* c1b  = (const float*)d_in[22];
  const float* c2_w = (const float*)d_in[23]; const float* c2_b = (const float*)d_in[24];
  const float* c21_w= (const float*)d_in[25]; const float* c21_b= (const float*)d_in[26];
  const float* c3_w = (const float*)d_in[27]; const float* c3_b = (const float*)d_in[28];
  const float* c31_w= (const float*)d_in[29]; const float* c31_b= (const float*)d_in[30];
  const float* g1g  = (const float*)d_in[31]; const float* g1gb = (const float*)d_in[32];
  const float* g2g  = (const float*)d_in[33]; const float* g2gb = (const float*)d_in[34];
  const float* g1z  = (const float*)d_in[35]; const float* g1zb = (const float*)d_in[36];
  const float* g2z  = (const float*)d_in[37]; const float* g2zb = (const float*)d_in[38];
  const float* g3z  = (const float*)d_in[39]; const float* g3zb = (const float*)d_in[40];
  const float* gz   = (const float*)d_in[41]; const float* gzb  = (const float*)d_in[42];
  const float* st1_w= (const float*)d_in[43]; const float* st1_b= (const float*)d_in[44];
  const float* st2_w= (const float*)d_in[45]; const float* st2_b= (const float*)d_in[46];
  const float* st3_w= (const float*)d_in[47]; const float* st3_b= (const float*)d_in[48];
  const float* e_w  = (const float*)d_in[49]; const float* e_b  = (const float*)d_in[50];
  const float* z0   = (const float*)d_in[51];

  // ---- workspace carve (f32 region, then f16 region) ----
  float* Xp  = (float*)d_ws;        // 16,384,000 f  [T,B,4H]
  float* sqB = Xp + 16384000;       //    256,000 f  [T,B,S]
  float* gB  = sqB + 256000;        //     32,768 f  [B,4H]
  float* cSt = gB + 32768;          //      8,192 f
  _Float16* x16    = (_Float16*)(cSt + 8192);  // 8,192,000 h [T,B,D]
  _Float16* hs0_16 = x16 + 8192000;            // 4,096,000 h
  _Float16* hs1_16 = hs0_16 + 4096000;         // 4,096,000 h
  _Float16* zq16   = hs1_16 + 4096000;         // 4,096,000 h
  _Float16* hSt16  = zq16 + 4096000;           //     8,192 h
  _Float16* ring16 = hSt16 + 8192;             //    20,480 h [5,B,Z]
  _Float16* p      = ring16 + 20480;           // f16 weights
  _Float16* tA16 = x16;                        // reuse x16 region after layer-0 GEMM
  _Float16* tB16 = x16 + 4096000;

  _Float16* Wih0_16 = p; p += 512 * 256;
  _Float16* Whh0_16 = p; p += 512 * 128;
  _Float16* Wih1_16 = p; p += 512 * 128;
  _Float16* Whh1_16 = p; p += 512 * 128;
  _Float16* f1_16 = p; p += 128 * 128;
  _Float16* f2_16 = p; p += 128 * 128;
  _Float16* f3_16 = p; p += 128 * 128;
  _Float16* s1_16 = p; p += 128 * 128;
  _Float16* s2_16 = p; p += 128 * 128;
  _Float16* s3_16 = p; p += 64 * 128;    // padded N: 8 -> 64
  _Float16* c2_16 = p; p += 64 * 128;
  _Float16* c21_16 = p; p += 64 * 64;
  _Float16* c3_16 = p; p += 64 * 128;
  _Float16* c31_16 = p; p += 64 * 64;
  _Float16* ew_16 = p; p += 256 * 64;
  _Float16* st1_16 = p; p += 64 * 64;
  _Float16* st2_16 = p; p += 64 * 64;
  _Float16* c1T_16 = p; p += 3 * 128 * 64;   // [l][128n][64k] (transposed)
  _Float16* g1gT_16 = p; p += 24 * 64 * 64;  // [s*l][TDn][Zk] (transposed)
  _Float16* g1zT_16 = p; p += 24 * 64 * 64;
  _Float16* g2gT_16 = p; p += 8 * 64 * 64;   // [s][Zn][TDk] (transposed)
  _Float16* g2zT_16 = p; p += 8 * 64 * 64;
  _Float16* g3zT_16 = p; p += 8 * 64 * 64;
  _Float16* gzT_16  = p; p += 24 * 64 * 64;

  float* out  = (float*)d_out;
  float* Y    = out;             //  8,192,000 [T,B,D]
  float* PMU  = out + 8192000;   // 16,384,000 [T,S,B,Z]
  float* PSIG = out + 24576000;  // 16,384,000
  float* QS   = out + 40960000;  //    256,000 [T,B,S]
  float* PS   = out + 41216000;  //    256,000
  float* QZMU = out + 41472000;  //  2,048,000 [T,B,Z]
  float* QZSIG= out + 43520000;  //  2,048,000

  // ---- operand prep (f16 conversion / transposition / padding) ----
  cvt_f16<<<4096, 256, 0, stream>>>(x, x16, 8192000);
  prep_w<<<256, 256, 0, stream>>>(Wih0, Wih0_16, 1, 512, 256, 512, 0);
  prep_w<<<256, 256, 0, stream>>>(Whh0, Whh0_16, 1, 512, 128, 512, 0);
  prep_w<<<256, 256, 0, stream>>>(Wih1, Wih1_16, 1, 512, 128, 512, 0);
  prep_w<<<256, 256, 0, stream>>>(Whh1, Whh1_16, 1, 512, 128, 512, 0);
  prep_w<<<64, 256, 0, stream>>>(f1_w, f1_16, 1, 128, 128, 128, 0);
  prep_w<<<64, 256, 0, stream>>>(f2_w, f2_16, 1, 128, 128, 128, 0);
  prep_w<<<64, 256, 0, stream>>>(f3_w, f3_16, 1, 128, 128, 128, 0);
  prep_w<<<64, 256, 0, stream>>>(s1_w, s1_16, 1, 128, 128, 128, 0);
  prep_w<<<64, 256, 0, stream>>>(s2_w, s2_16, 1, 128, 128, 128, 0);
  prep_w<<<32, 256, 0, stream>>>(s3_w, s3_16, 1, 8, 128, 64, 0);
  prep_w<<<32, 256, 0, stream>>>(c2_w, c2_16, 1, 64, 128, 64, 0);
  prep_w<<<16, 256, 0, stream>>>(c21_w, c21_16, 1, 64, 64, 64, 0);
  prep_w<<<32, 256, 0, stream>>>(c3_w, c3_16, 1, 64, 128, 64, 0);
  prep_w<<<16, 256, 0, stream>>>(c31_w, c31_16, 1, 64, 64, 64, 0);
  prep_w<<<64, 256, 0, stream>>>(e_w, ew_16, 1, 256, 64, 256, 0);
  prep_w<<<16, 256, 0, stream>>>(st1_w, st1_16, 1, 64, 64, 64, 0);
  prep_w<<<16, 256, 0, stream>>>(st2_w, st2_16, 1, 64, 64, 64, 0);
  prep_w<<<96, 256, 0, stream>>>(c1, c1T_16, 3, 128, 64, 128, 1);
  prep_w<<<384, 256, 0, stream>>>(g1g, g1gT_16, 24, 64, 64, 64, 1);
  prep_w<<<384, 256, 0, stream>>>(g1z, g1zT_16, 24, 64, 64, 64, 1);
  prep_w<<<128, 256, 0, stream>>>(g2g, g2gT_16, 8, 64, 64, 64, 1);
  prep_w<<<128, 256, 0, stream>>>(g2z, g2zT_16, 8, 64, 64, 64, 1);
  prep_w<<<128, 256, 0, stream>>>(g3z, g3zT_16, 8, 64, 64, 64, 1);
  prep_w<<<384, 256, 0, stream>>>(gz, gzT_16, 24, 64, 64, 64, 1);

  // ---- LSTM layer 0 ----
  gemm_big<<<dim3(500, 8), 256, 0, stream>>>(x16, D_, Wih0_16, D_, 512,
                                             bih0, bhh0, Xp, nullptr, 512, D_, 0);
  lstm_seq<<<1, 256, 0, stream>>>(Xp, Whh0_16, hs0_16, hSt16, cSt, gB);
  // ---- LSTM layer 1 ----
  gemm_big<<<dim3(500, 8), 256, 0, stream>>>(hs0_16, H_, Wih1_16, H_, 512,
                                             bih1, bhh1, Xp, nullptr, 512, H_, 0);
  lstm_seq<<<1, 256, 0, stream>>>(Xp, Whh1_16, hs1_16, hSt16, cSt, gB);
  // ---- heads ----
  gemm_big<<<dim3(500, 2), 256, 0, stream>>>(hs1_16, H_, f1_16, H_, 128,
                                             f1_b, nullptr, nullptr, tA16, 128, H_, 1);
  gemm_big<<<dim3(500, 2), 256, 0, stream>>>(tA16, H_, f2_16, H_, 128,
                                             f2_b, nullptr, nullptr, tB16, 128, H_, 1);
  gemm_big<<<dim3(500, 2), 256, 0, stream>>>(tB16, H_, f3_16, H_, 128,
                                             f3_b, nullptr, nullptr, zq16, 128, H_, 0);
  gemm_big<<<dim3(500, 2), 256, 0, stream>>>(hs1_16, H_, s1_16, H_, 128,
                                             s1_b, nullptr, nullptr, tA16, 128, H_, 1);
  gemm_big<<<dim3(500, 2), 256, 0, stream>>>(tA16, H_, s2_16, H_, 128,
                                             s2_b, nullptr, nullptr, tB16, 128, H_, 1);
  gemm_big<<<dim3(500, 1), 256, 0, stream>>>(tB16, H_, s3_16, H_, S_,
                                             s3_b, nullptr, sqB, nullptr, S_, H_, 0);
  // ---- lag ring init ----
  init_buf<<<(MAXL_ * B_ * Z_ + 255) / 256, 256, 0, stream>>>(z0, ring16);
  // ---- sequential scan over time ----
  for (int t = 0; t < T_; ++t) {
    step_par<<<S_ + 1, 256, 0, stream>>>(t, ring16, sqB,
        g1gT_16, g1gb, g2gT_16, g2gb, g1zT_16, g1zb, g2zT_16, g2zb,
        g3zT_16, g3zb, gzT_16, gzb,
        st1_16, st1_b, st2_16, st2_b, st3_w, st3_b,
        PMU, PSIG, QS, PS);
    step_comb<<<1, 256, 0, stream>>>(t, ring16, zq16,
        c1T_16, c1b, c2_16, c2_b, c21_16, c21_b, c3_16, c3_b, c31_16, c31_b,
        ew_16, e_b, Y, QZMU, QZSIG);
  }
}
